// RWKVAttention_28381143892187
// MI455X (gfx1250) — matrix operand order repaired and checked by reading, not compiled, not run on hardware
//
#include <hip/hip_runtime.h>

typedef __bf16 bf16_t;
typedef __attribute__((ext_vector_type(16))) __bf16 v16bf;
typedef __attribute__((ext_vector_type(8)))  __bf16 v8bf;
typedef __attribute__((ext_vector_type(8)))  float  v8f;

#define B_    2
#define S_    4096
#define HID_  1024
#define H_    16
#define M_    (B_ * S_)      // 8192 rows
#define N1_   (3 * HID_)     // 3072
#define K_    HID_           // 1024
#define NCH   16
#define CHUNK (S_ / NCH)     // 256

#define TM 128
#define TN 128
#define TK 32
#define LDR 40               // TK + 8 bf16 pad (80B rows -> bank spread, 16B aligned)

static __device__ __forceinline__ bf16_t f2bf(float f) {
  union { unsigned int u; float f; } in; in.f = f;
  unsigned int x = in.u;
  x += 0x7FFFu + ((x >> 16) & 1u);        // round-to-nearest-even
  union { unsigned short s; bf16_t b; } out;
  out.s = (unsigned short)(x >> 16);
  return out.b;
}

__global__ void cvt_f32_bf16(const float* __restrict__ src, bf16_t* __restrict__ dst, int n) {
  int i = blockIdx.x * blockDim.x + threadIdx.x;
  int stride = gridDim.x * blockDim.x;
  for (; i < n; i += stride) dst[i] = f2bf(src[i]);
}

// mixed = hs*tm + shift(hs)*(1-tm), cast to bf16
__global__ void mix_kernel(const float* __restrict__ hs, const float* __restrict__ tm,
                           bf16_t* __restrict__ dst) {
  int idx = blockIdx.x * blockDim.x + threadIdx.x;   // over B*S*HID (exact)
  int h = idx & (HID_ - 1);
  int s = (idx >> 10) & (S_ - 1);
  float cur  = hs[idx];
  float prev = (s > 0) ? hs[idx - HID_] : 0.0f;
  float t = tm[h];
  dst[idx] = f2bf(cur * t + prev * (1.0f - t));
}

__global__ void avg_decay_kernel(const float* __restrict__ td, float* __restrict__ out) {
  __shared__ float red[1024];
  int t = threadIdx.x;
  red[t] = -__expf(td[t]);
  __syncthreads();
  for (int s = 512; s > 0; s >>= 1) {
    if (t < s) red[t] += red[t + s];
    __syncthreads();
  }
  if (t == 0) out[0] = red[0] / 1024.0f;
}

// D[m][n] = sum_k A[m][k] * Bw[n][k]   (A: MxK row-major bf16, Bw: NxK row-major bf16)
// Double-buffered LDS pipeline: one barrier per K-step, global prefetch overlapped
// with the 8 WMMAs of the current buffer.
__global__ void __launch_bounds__(256)
gemm_bf16_nt(const bf16_t* __restrict__ A, const bf16_t* __restrict__ Bw,
             float* __restrict__ Dm, int N, int K) {
  __shared__ bf16_t As[2][TM * LDR];
  __shared__ bf16_t Bs[2][TN * LDR];

  const int m0 = blockIdx.y * TM;
  const int n0 = blockIdx.x * TN;
  const int t    = threadIdx.x;
  const int lane = t & 31;
  const int wv   = t >> 5;             // 8 waves: 2 (M) x 4 (N)
  const int wm   = (wv >> 2) * 64;
  const int wn   = (wv & 3) * 32;
  const int lm   = lane & 15;
  const int lk8  = (lane >> 4) * 8;    // A frag K sub-offset
  const int lk16 = (lane >> 4) * 16;   // unused (B frag uses lk8 too)

  const int srow = t >> 1;             // staging: 128 rows x 2 halves
  const int scol = (t & 1) * 16;
  const int soff = srow * LDR + scol;

  const bf16_t* gA = A + (size_t)(m0 + srow) * K + scol;
  const bf16_t* gB = Bw + (size_t)(n0 + srow) * K + scol;

  v8f acc[4][2] = {};

  // preload first tile
  {
    v8bf a0 = *(const v8bf*)(gA);
    v8bf a1 = *(const v8bf*)(gA + 8);
    v8bf b0 = *(const v8bf*)(gB);
    v8bf b1 = *(const v8bf*)(gB + 8);
    *(v8bf*)(&As[0][soff])     = a0;
    *(v8bf*)(&As[0][soff + 8]) = a1;
    *(v8bf*)(&Bs[0][soff])     = b0;
    *(v8bf*)(&Bs[0][soff + 8]) = b1;
  }
  __syncthreads();

  int buf = 0;
  for (int kt = 0; kt < K; kt += TK) {
    const bool more = (kt + TK) < K;
    v8bf na0, na1, nb0, nb1;
    if (more) {                        // issue next global loads before compute
      na0 = *(const v8bf*)(gA + kt + TK);
      na1 = *(const v8bf*)(gA + kt + TK + 8);
      nb0 = *(const v8bf*)(gB + kt + TK);
      nb1 = *(const v8bf*)(gB + kt + TK + 8);
    }

    v16bf bfr[2];
#pragma unroll
    for (int j = 0; j < 2; ++j) {
      const bf16_t* p = &Bs[buf][(wn + j * 16 + lm) * LDR + lk8];
      v8bf lo = *(const v8bf*)(p);
      v8bf hi = *(const v8bf*)(p + 16);
#pragma unroll
      for (int q = 0; q < 8; ++q) { bfr[j][q] = lo[q]; bfr[j][q + 8] = hi[q]; }
    }
#pragma unroll
    for (int i = 0; i < 4; ++i) {
      const bf16_t* p = &As[buf][(wm + i * 16 + lm) * LDR];
      v8bf lo = *(const v8bf*)(p + lk8);
      v8bf hi = *(const v8bf*)(p + 16 + lk8);
      v16bf af;
#pragma unroll
      for (int q = 0; q < 8; ++q) { af[q] = lo[q]; af[q + 8] = hi[q]; }
#pragma unroll
      for (int j = 0; j < 2; ++j) {
        acc[i][j] = __builtin_amdgcn_wmma_f32_16x16x32_bf16(
            false, af, false, bfr[j], (short)0, acc[i][j], false, false);
      }
    }

    if (more) {                        // stage into the other buffer, single barrier
      const int nb = buf ^ 1;
      *(v8bf*)(&As[nb][soff])     = na0;
      *(v8bf*)(&As[nb][soff + 8]) = na1;
      *(v8bf*)(&Bs[nb][soff])     = nb0;
      *(v8bf*)(&Bs[nb][soff + 8]) = nb1;
      __syncthreads();
    }
    buf ^= 1;
  }

  // epilogue: 32-bit index math (max index 8192*3072 < 2^31)
  float* Dp = Dm + (unsigned)(m0 + wm + (lane >> 4) * 8) * (unsigned)N
                 + (unsigned)(n0 + wn + lm);
#pragma unroll
  for (int i = 0; i < 4; ++i)
#pragma unroll
    for (int j = 0; j < 2; ++j)
#pragma unroll
      for (int v = 0; v < 8; ++v)
        Dp[(unsigned)(i * 16 + v) * (unsigned)N + (unsigned)(j * 16)] = acc[i][j][v];
}

// sigmoid(r), RoPE on r and k, kv = k*v; written in place over rkv sections 0 (r) and 1 (kv)
__global__ void rope_kv_kernel(float* __restrict__ rkv,
                               const float* __restrict__ cosb,
                               const float* __restrict__ sinb) {
  int tid = blockIdx.x * blockDim.x + threadIdx.x;   // B*S*H*32 (pairs d, d+32)
  int d = tid & 31;
  int h = (tid >> 5) & (H_ - 1);
  int s = (tid >> 9) & (S_ - 1);
  int b = tid >> 21;
  size_t base = (size_t)(b * S_ + s) * 3072 + h * 64;
  float r0 = rkv[base + d],        r1 = rkv[base + d + 32];
  float k0 = rkv[base + 1024 + d], k1 = rkv[base + 1024 + d + 32];
  float v0 = rkv[base + 2048 + d], v1 = rkv[base + 2048 + d + 32];
  r0 = 1.0f / (1.0f + __expf(-r0));
  r1 = 1.0f / (1.0f + __expf(-r1));
  float c0 = cosb[s * 64 + d], c1 = cosb[s * 64 + d + 32];
  float n0 = sinb[s * 64 + d], n1 = sinb[s * 64 + d + 32];
  float rr0 = r0 * c0 - r1 * n0;
  float rr1 = r1 * c1 + r0 * n1;
  float kk0 = k0 * c0 - k1 * n0;
  float kk1 = k1 * c1 + k0 * n1;
  rkv[base + d]             = rr0;
  rkv[base + d + 32]        = rr1;
  rkv[base + 1024 + d]      = kk0 * v0;   // kv overwrites k section
  rkv[base + 1024 + d + 32] = kk1 * v1;
}

// Pass A: per (b, chunk, c) local suffix sum  T = sum_{s in chunk} lam^{s-s0} kv[s]
__global__ void scanA_kernel(const float* __restrict__ rkv, const float* __restrict__ avg,
                             float* __restrict__ Tbuf) {
  int tid = blockIdx.x * blockDim.x + threadIdx.x;   // B*NCH*1024
  int c  = tid & 1023;
  int ch = (tid >> 10) & (NCH - 1);
  int b  = tid >> 14;
  float g   = -avg[0];                 // dw_unnorm[i][j] = exp(g*(j-i)), j>=i
  float lam = __expf(g);
  int s0 = ch * CHUNK;
  float acc = 0.0f;
  for (int s = s0 + CHUNK - 1; s >= s0; --s) {
    float kv = rkv[(size_t)(b * S_ + s) * 3072 + 1024 + c];
    acc = fmaf(lam, acc, kv);
  }
  Tbuf[(b * NCH + ch) * 1024 + c] = acc;
}

// Pass B: propagate lam^CHUNK carry, emit attn = r * (tf*kv + suffix/nrm) as bf16
__global__ void scanB_kernel(const float* __restrict__ rkv, const float* __restrict__ avg,
                             const float* __restrict__ Tbuf,
                             const float* __restrict__ tf,
                             bf16_t* __restrict__ attn) {
  int tid = blockIdx.x * blockDim.x + threadIdx.x;
  int c  = tid & 1023;
  int ch = (tid >> 10) & (NCH - 1);
  int b  = tid >> 14;
  float g    = -avg[0];
  float lam  = __expf(g);
  float lamC = __expf(g * (float)CHUNK);
  float acc = 0.0f;
  for (int q = NCH - 1; q > ch; --q)
    acc = fmaf(lamC, acc, Tbuf[(b * NCH + q) * 1024 + c]);
  int s0 = ch * CHUNK;
  int s1 = s0 + CHUNK;
  // nrm(s) = (lam^(S-s) - 1)/(lam - 1); recurrence nrm_s = 1 + lam*nrm_{s+1}
  float nrm = (__expf(g * (float)(S_ - s1)) - 1.0f) / (lam - 1.0f);
  float tfc = tf[c];
  for (int s = s1 - 1; s >= s0; --s) {
    size_t o = (size_t)(b * S_ + s) * 3072;
    float kv = rkv[o + 1024 + c];
    acc = fmaf(lam, acc, kv);
    nrm = fmaf(lam, nrm, 1.0f);
    float wkv = acc / (nrm + 1e-8f);
    float r = rkv[o + c];
    attn[(size_t)(b * S_ + s) * 1024 + c] = f2bf(r * (tfc * kv + wkv));
  }
}

extern "C" void kernel_launch(void* const* d_in, const int* in_sizes, int n_in,
                              void* d_out, int out_size, void* d_ws, size_t ws_size,
                              hipStream_t stream) {
  (void)in_sizes; (void)n_in; (void)out_size; (void)ws_size;
  const float* hs   = (const float*)d_in[0];
  const float* cosb = (const float*)d_in[1];
  const float* sinb = (const float*)d_in[2];
  const float* Wrkv = (const float*)d_in[3];
  const float* Wo   = (const float*)d_in[4];
  const float* td   = (const float*)d_in[5];
  const float* tf   = (const float*)d_in[6];
  const float* tm   = (const float*)d_in[7];
  float* out = (float*)d_out;

  char* ws = (char*)d_ws;
  size_t off = 0;
  auto alloc = [&](size_t bytes) {
    char* p = ws + off;
    off += (bytes + 255) & ~(size_t)255;
    return p;
  };
  bf16_t* wrkv_bf  = (bf16_t*)alloc((size_t)N1_ * K_ * 2);   //  6.3 MB
  bf16_t* wo_bf    = (bf16_t*)alloc((size_t)HID_ * K_ * 2);  //  2.1 MB
  bf16_t* mixed_bf = (bf16_t*)alloc((size_t)M_ * K_ * 2);    // 16.8 MB
  float*  rkv      = (float*) alloc((size_t)M_ * N1_ * 4);   // 100.7 MB
  bf16_t* attn_bf  = (bf16_t*)alloc((size_t)M_ * HID_ * 2);  // 16.8 MB
  float*  avg      = (float*) alloc(256);
  float*  Tbuf     = (float*) alloc((size_t)B_ * NCH * 1024 * 4);

  cvt_f32_bf16<<<2048, 256, 0, stream>>>(Wrkv, wrkv_bf, N1_ * K_);
  cvt_f32_bf16<<<1024, 256, 0, stream>>>(Wo, wo_bf, HID_ * K_);
  mix_kernel<<<(M_ * K_) / 256, 256, 0, stream>>>(hs, tm, mixed_bf);
  avg_decay_kernel<<<1, 1024, 0, stream>>>(td, avg);

  dim3 g1(N1_ / TN, M_ / TM);
  gemm_bf16_nt<<<g1, 256, 0, stream>>>(mixed_bf, wrkv_bf, rkv, N1_, K_);

  rope_kv_kernel<<<(B_ * S_ * H_ * 32) / 256, 256, 0, stream>>>(rkv, cosb, sinb);
  scanA_kernel<<<(B_ * NCH * 1024) / 256, 256, 0, stream>>>(rkv, avg, Tbuf);
  scanB_kernel<<<(B_ * NCH * 1024) / 256, 256, 0, stream>>>(rkv, avg, Tbuf, tf, attn_bf);

  dim3 g2(HID_ / TN, M_ / TM);
  gemm_bf16_nt<<<g2, 256, 0, stream>>>(attn_bf, wo_bf, out, HID_, K_);
}
